// TaylorFeatureMap_27496380629315
// MI455X (gfx1250) — compile-verified, matrix operand order checked by reading, not run on hardware
//
#include <hip/hip_runtime.h>

// Taylor feature map for MI455X (gfx1250, wave32).
// One wave (= one 32-thread block) per 64-length row; row base address is
// scalar -> saddr-form stores. Outer product via rank-1 V_WMMA_F32_16X16X4_F32
// (10 upper tiles). Diagonal = x^2 computed elementwise (no WMMA needed).
// B operand pre-scaled by 1/sqrt(D) so WMMA outputs are store-ready.
// Output-bandwidth bound: ~562 MB of NT writes -> ~25 us floor at 23.3 TB/s.

typedef float v2f __attribute__((ext_vector_type(2)));
typedef float v8f __attribute__((ext_vector_type(8)));
typedef _Float16 v16h __attribute__((ext_vector_type(16)));

#define HEAD_DIM    64
#define OUT_PER_ROW 2145   // 1 + 64 + 64 + 64*63/2

// Rank-1 16x16 tile: D[m][n] = am * bn.
//  - am = x[cm*16 + lane] on lanes 0..15, 0.0f on lanes 16..31 (zeros in the
//    K=1..3 slots of the 16x4 f32 A layout kill every K except K=0).
//  - bn replicated into all B slots (N = lane&15 in every B layout); A's
//    zero K-slots make the replicas harmless.
__device__ __forceinline__ v8f rank1_tile(float am_or_zero, float bn) {
#if __has_builtin(__builtin_amdgcn_wmma_f32_16x16x4_f32)
  v2f a; a.x = am_or_zero; a.y = 0.0f;
  v2f b; b.x = bn;         b.y = bn;
  v8f c = {};
  return __builtin_amdgcn_wmma_f32_16x16x4_f32(
      /*neg_a=*/false, a, /*neg_b=*/false, b,
      /*c_mod=*/(short)0, c, /*reuse_a=*/false, /*reuse_b=*/false);
#else
  v16h a = {};
  a[0] = (_Float16)am_or_zero;
  v16h b;
  _Float16 h = (_Float16)bn;
  #pragma unroll
  for (int k = 0; k < 16; ++k) b[k] = h;
  v8f c = {};
  return __builtin_amdgcn_wmma_f32_16x16x32_f16(
      false, a, false, b, (short)0, c, false, false);
#endif
}

__global__ void __launch_bounds__(32)
taylor_fm_kernel(const float* __restrict__ x, float* __restrict__ out) {
  const int lane = threadIdx.x;          // 0..31 (one wave per block)
  const int row  = blockIdx.x;           // uniform -> SGPR base address

  const float inv_rrd   = 0.35355339059327373f;   // 1 / 64^0.25
  const float inv_rd    = 0.125f;                 // 1 / 8
  const float inv_rd_r2 = 0.08838834764831845f;   // 1 / (8*sqrt(2))

  const float* __restrict__ xr = x + (size_t)row * HEAD_DIM;
  float* __restrict__ o = out + (size_t)row * OUT_PER_ROW;

  // Coalesced row load.
  const float xa = xr[lane];
  const float xb = xr[lane + 32];

  // Sections [0..128]: 1, x/ D^0.25, diag(x x^T)/(sqrt(D)*sqrt(2)) = x^2 * c.
  if (lane == 0) __builtin_nontemporal_store(1.0f, o);
  __builtin_nontemporal_store(xa * inv_rrd,        o + 1  + lane);
  __builtin_nontemporal_store(xb * inv_rrd,        o + 33 + lane);
  __builtin_nontemporal_store(xa * xa * inv_rd_r2, o + 65 + lane);
  __builtin_nontemporal_store(xb * xb * inv_rd_r2, o + 97 + lane);

  const int n    = lane & 15;
  const int half = lane >> 4;
  const int tbl  = n - 8 * half;          // j - i0 for same-chunk tiles

  // v[c] = x[c*16 + n] in every lane (ds_bpermute).
  float v[4];
  v[0] = __shfl(xa, n,      32);
  v[1] = __shfl(xa, 16 + n, 32);
  v[2] = __shfl(xb, n,      32);
  v[3] = __shfl(xb, 16 + n, 32);

  // Pre-scaled B operands: WMMA result = x[i] * x[j] / sqrt(D), store-ready.
  float bs[4];
  #pragma unroll
  for (int c = 0; c < 4; ++c) bs[c] = v[c] * inv_rd;

  char* ob = (char*)o;

  // ---- Diagonal-chunk tiles (cm == cn): strict upper part only. ----------
  // All 4 WMMAs first, then 8 shared predicate regions (pred: tbl > r).
  v8f  dd[4];
  int  db[4], dssl[4];
  #pragma unroll
  for (int c = 0; c < 4; ++c) {
    const float am = half ? 0.0f : v[c];
    dd[c] = rank1_tile(am, bs[c]);
    const int i0lo = c * 16, i0hi = c * 16 + 8;
    const int tri_lo = i0lo * (127 - i0lo) / 2;   // compile-time consts;
    const int tri_hi = i0hi * (127 - i0hi) / 2;   // runtime = one cndmask
    // byte offset of element (i0, j) in packed triu: 4*(128 + tri(i0) + tbl)
    db[c]   = ((half ? tri_hi : tri_lo) + 128 + tbl) << 2;
    dssl[c] = (half ? (54 - c * 16) : (62 - c * 16)) << 2;  // 4*(62 - i0)
  }
  #pragma unroll
  for (int r = 0; r < 8; ++r) {
    if (tbl > r) {   // j > i  (one exec region covers all 4 tiles)
      #pragma unroll
      for (int c = 0; c < 4; ++c) {
        const int br = db[c] + r * dssl[c] - 2 * r * (r - 1);  // v_mad + imm
        __builtin_nontemporal_store(dd[c][r], (float*)(ob + br));
      }
    }
  }

  // ---- Full upper tiles (cm < cn): unconditional stores. -----------------
  #pragma unroll
  for (int cm = 0; cm < 4; ++cm) {
    #pragma unroll
    for (int cn = cm + 1; cn < 4; ++cn) {
      const float am = half ? 0.0f : v[cm];
      v8f d = rank1_tile(am, bs[cn]);
      const int i0lo = cm * 16, i0hi = cm * 16 + 8;
      const int tri_lo = i0lo * (127 - i0lo) / 2;
      const int tri_hi = i0hi * (127 - i0hi) / 2;
      const int b = ((half ? tri_hi : tri_lo) + 128 + (cn - cm) * 16 + tbl) << 2;
      const int s = (half ? (54 - cm * 16) : (62 - cm * 16)) << 2;
      #pragma unroll
      for (int r = 0; r < 8; ++r) {
        const int br = b + r * s - 2 * r * (r - 1);
        __builtin_nontemporal_store(d[r], (float*)(ob + br));
      }
    }
  }
}

extern "C" void kernel_launch(void* const* d_in, const int* in_sizes, int n_in,
                              void* d_out, int out_size, void* d_ws, size_t ws_size,
                              hipStream_t stream) {
  const float* x = (const float*)d_in[0];
  float* out = (float*)d_out;
  const int nrows = in_sizes[0] / HEAD_DIM;   // 2*16*2048 = 65536
  taylor_fm_kernel<<<nrows, 32, 0, stream>>>(x, out);
}
